// MixtureOfMambaBlock_81853486727573
// MI455X (gfx1250) — compile-verified, matrix operand order checked by reading, not run on hardware
//
#include <hip/hip_runtime.h>
#include <hip/hip_bf16.h>

// ---------------- problem constants ----------------
#define BB   2
#define LL   1024
#define DD   1024
#define EE   8
#define TOPK 2
#define NN   16
#define DI   2048
#define RR   64
#define KW   4
#define ML   (BB * LL)          // 2048 rows
#define XZW  (2 * DI)           // 4096
#define PROJW (RR + 2 * NN)     // 96

typedef __attribute__((ext_vector_type(16))) __bf16 v16bf;
typedef __attribute__((ext_vector_type(8)))  __bf16 v8bf;
typedef __attribute__((ext_vector_type(8)))  float  v8f;

// ---------------- helpers ----------------
__device__ __forceinline__ float silu_f(float x) { return x / (1.0f + __expf(-x)); }
__device__ __forceinline__ float softplus_f(float x) {
    return (x > 20.0f) ? x : log1pf(__expf(x));
}

// Load one 16x32 bf16 operand fragment (A or B^T), K-major rows, no guard.
// Layout per CDNA5 ISA 7.12.2: lanes 0-15 row r hold K 0..7,16..23;
// lanes 16-31 hold K 8..15,24..31.
__device__ __forceinline__ v16bf load_frag_nc(const __hip_bfloat16* base, int ld) {
    int lane = threadIdx.x & 31;
    const __bf16* p = reinterpret_cast<const __bf16*>(base) +
                      (long)(lane & 15) * ld + (lane >> 4) * 8;
    v8bf lo = *reinterpret_cast<const v8bf*>(p);
    v8bf hi = *reinterpret_cast<const v8bf*>(p + 16);
    return __builtin_shufflevector(lo, hi, 0, 1, 2, 3, 4, 5, 6, 7,
                                   8, 9, 10, 11, 12, 13, 14, 15);
}

// Guarded variant (rows may exceed matrix bound)
__device__ __forceinline__ v16bf load_frag_g(const __hip_bfloat16* base, int ld,
                                             int valid_rows) {
    int lane = threadIdx.x & 31;
    int row  = lane & 15;
    v8bf lo = {};
    v8bf hi = {};
    if (row < valid_rows) {
        const __bf16* p = reinterpret_cast<const __bf16*>(base) +
                          (long)row * ld + (lane >> 4) * 8;
        lo = *reinterpret_cast<const v8bf*>(p);
        hi = *reinterpret_cast<const v8bf*>(p + 16);
    }
    return __builtin_shufflevector(lo, hi, 0, 1, 2, 3, 4, 5, 6, 7,
                                   8, 9, 10, 11, 12, 13, 14, 15);
}

// ---------------- kernels ----------------
__global__ void k_zero(float* p, long n) {
    long i = (long)blockIdx.x * blockDim.x + threadIdx.x;
    if (i < n) p[i] = 0.0f;
}

__global__ void k_f32_to_bf16(const float* __restrict__ src,
                              __hip_bfloat16* __restrict__ dst, long n) {
    long i = (long)blockIdx.x * blockDim.x + threadIdx.x;
    if (i < n) dst[i] = __float2bfloat16(src[i]);
}

// copy src[r*src_ld + c] (c<cols) -> dst[r*cols + c] as bf16
__global__ void k_slice_to_bf16(const float* __restrict__ src,
                                __hip_bfloat16* __restrict__ dst,
                                int rows, int src_ld, int cols) {
    long i = (long)blockIdx.x * blockDim.x + threadIdx.x;
    long n = (long)rows * cols;
    if (i >= n) return;
    int c = (int)(i % cols);
    int r = (int)(i / cols);
    dst[i] = __float2bfloat16(src[(long)r * src_ld + c]);
}

// Router: logits = x.gate_w^T, top-2 softmax, dense combine weights,
// load + entropy accumulators.
__global__ void k_router(const float* __restrict__ x,
                         const float* __restrict__ gate_w,
                         float* __restrict__ weights,
                         float* __restrict__ load_acc,
                         float* __restrict__ ent_acc) {
    int p = blockIdx.x * blockDim.x + threadIdx.x;
    if (p >= ML) return;
    float logit[EE];
    const float* xp = x + (long)p * DD;
#pragma unroll
    for (int e = 0; e < EE; ++e) {
        const float* g = gate_w + (long)e * DD;
        float s = 0.0f;
        for (int i = 0; i < DD; ++i) s = fmaf(xp[i], g[i], s);
        logit[e] = s;
    }
    int i1 = 0;
#pragma unroll
    for (int e = 1; e < EE; ++e) if (logit[e] > logit[i1]) i1 = e;
    int i2 = (i1 == 0) ? 1 : 0;
#pragma unroll
    for (int e = 0; e < EE; ++e)
        if (e != i1 && logit[e] > logit[i2]) i2 = e;
    float m  = logit[i1];
    float e1 = __expf(logit[i1] - m);
    float e2 = __expf(logit[i2] - m);
    float s  = e1 + e2;
    float g1 = e1 / s, g2 = e2 / s;
#pragma unroll
    for (int e = 0; e < EE; ++e) weights[(long)p * EE + e] = 0.0f;
    weights[(long)p * EE + i1] = g1;
    weights[(long)p * EE + i2] = g2;
    atomicAdd(&load_acc[i1], g1);
    atomicAdd(&load_acc[i2], g2);
    float ent = -(g1 * __logf(fmaxf(g1, 1e-30f)) + g2 * __logf(fmaxf(g2, 1e-30f)));
    atomicAdd(ent_acc, ent);
}

// ----- Full-tile WMMA GEMM: M%256==0, N%128==0, K%32==0 -----
// out[M,N] = A[M,K](bf16) * W[N,K](bf16)^T, f32 accumulate.
// 256 threads = 8 waves (4 in M x 2 in N); wave tile 64x64 -> 16 WMMA / K-step.
// mode 0: store; mode 1: softplus(v + bias[n]); mode 2: out += rowscale[m*rs]*v
__global__ void __launch_bounds__(256)
k_wmma_gemm_full(const __hip_bfloat16* __restrict__ A,
                 const __hip_bfloat16* __restrict__ W,
                 float* __restrict__ out,
                 int M, int N, int K, int mode,
                 const float* __restrict__ bias,
                 const float* __restrict__ rowscale, int rs_stride) {
    const int wave = threadIdx.x >> 5;          // 0..7
    const int wm = wave >> 1;                   // 0..3
    const int wn = wave & 1;                    // 0..1
    const int m0 = blockIdx.x * 256 + wm * 64;
    const int n0 = blockIdx.y * 128 + wn * 64;

    v8f acc[4][4];
#pragma unroll
    for (int i = 0; i < 4; ++i)
#pragma unroll
        for (int j = 0; j < 4; ++j) acc[i][j] = (v8f){};

    const __hip_bfloat16* Ap = A + (long)m0 * K;
    const __hip_bfloat16* Wp = W + (long)n0 * K;

    for (int k0 = 0; k0 < K; k0 += 32) {
        if (k0 + 32 < K) {
            __builtin_prefetch(Ap + k0 + 32, 0, 1);
            __builtin_prefetch(Wp + k0 + 32, 0, 1);
        }
        v16bf a[4], b[4];
#pragma unroll
        for (int i = 0; i < 4; ++i)
            a[i] = load_frag_nc(Ap + (long)(16 * i) * K + k0, K);
#pragma unroll
        for (int j = 0; j < 4; ++j)
            b[j] = load_frag_nc(Wp + (long)(16 * j) * K + k0, K);
#pragma unroll
        for (int i = 0; i < 4; ++i)
#pragma unroll
            for (int j = 0; j < 4; ++j)
                acc[i][j] = __builtin_amdgcn_wmma_f32_16x16x32_bf16(
                    false, a[i], false, b[j], (short)0, acc[i][j], false, false);
    }

    const int lane = threadIdx.x & 31;
    const int col  = lane & 15;
    const int rb   = (lane >> 4) * 8;
#pragma unroll
    for (int i = 0; i < 4; ++i) {
#pragma unroll
        for (int j = 0; j < 4; ++j) {
            int n = n0 + 16 * j + col;
#pragma unroll
            for (int r = 0; r < 8; ++r) {
                int mm = m0 + 16 * i + rb + r;
                float v = acc[i][j][r];
                long idx = (long)mm * N + n;
                if (mode == 0) {
                    out[idx] = v;
                } else if (mode == 1) {
                    out[idx] = softplus_f(v + bias[n]);
                } else {
                    out[idx] += rowscale[(long)mm * rs_stride] * v;
                }
            }
        }
    }
}

// ----- Guarded WMMA GEMM (edge tiles allowed): 128 threads, wave tile 64x32 -----
__global__ void __launch_bounds__(128)
k_wmma_gemm_guard(const __hip_bfloat16* __restrict__ A,
                  const __hip_bfloat16* __restrict__ W,
                  float* __restrict__ out,
                  int M, int N, int K) {
    const int wave = threadIdx.x >> 5;
    const int wm = wave >> 1;
    const int wn = wave & 1;
    const int m0 = blockIdx.x * 128 + wm * 64;
    const int n0 = blockIdx.y * 64 + wn * 32;

    v8f acc[4][2];
#pragma unroll
    for (int i = 0; i < 4; ++i)
#pragma unroll
        for (int j = 0; j < 2; ++j) acc[i][j] = (v8f){};

    for (int k0 = 0; k0 < K; k0 += 32) {
        v16bf a[4], b[2];
#pragma unroll
        for (int i = 0; i < 4; ++i) {
            int mr = m0 + 16 * i;
            a[i] = load_frag_g(A + (long)mr * K + k0, K, M - mr);
        }
#pragma unroll
        for (int j = 0; j < 2; ++j) {
            int nr = n0 + 16 * j;
            b[j] = load_frag_g(W + (long)nr * K + k0, K, N - nr);
        }
#pragma unroll
        for (int i = 0; i < 4; ++i)
#pragma unroll
            for (int j = 0; j < 2; ++j)
                acc[i][j] = __builtin_amdgcn_wmma_f32_16x16x32_bf16(
                    false, a[i], false, b[j], (short)0, acc[i][j], false, false);
    }

    const int lane = threadIdx.x & 31;
    const int col  = lane & 15;
    const int rb   = (lane >> 4) * 8;
#pragma unroll
    for (int i = 0; i < 4; ++i) {
#pragma unroll
        for (int j = 0; j < 2; ++j) {
            int n = n0 + 16 * j + col;
            if (n >= N) continue;
#pragma unroll
            for (int r = 0; r < 8; ++r) {
                int mm = m0 + 16 * i + rb + r;
                if (mm >= M) continue;
                out[(long)mm * N + n] = acc[i][j][r];
            }
        }
    }
}

// depthwise causal conv (width 4) + SiLU; writes f32 and bf16 copies.
__global__ void k_conv_silu(const float* __restrict__ xz,
                            const float* __restrict__ convw,
                            const float* __restrict__ convb,
                            float* __restrict__ xc,
                            __hip_bfloat16* __restrict__ xcbf) {
    long i = (long)blockIdx.x * blockDim.x + threadIdx.x;
    if (i >= (long)ML * DI) return;
    int d = (int)(i % DI);
    long bl = i / DI;
    int l = (int)(bl % LL);
    long brow = bl - l;                       // b*L
    float s = convb[d];
#pragma unroll
    for (int k = 0; k < KW; ++k) {
        int lt = l + k - (KW - 1);
        if (lt >= 0)
            s = fmaf(convw[(long)d * KW + k], xz[(brow + lt) * XZW + d], s);
    }
    float v = silu_f(s);
    xc[i] = v;
    xcbf[i] = __float2bfloat16(v);
}

// selective scan over L: one thread per (b,d) channel, N=16 states in regs,
// B_t/C_t staged in LDS per timestep.  Output gated y in bf16.
__global__ void k_scan(const float* __restrict__ dt,
                       const float* __restrict__ proj,
                       const float* __restrict__ xc,
                       const float* __restrict__ xz,
                       const float* __restrict__ Alog,
                       const float* __restrict__ Dvec,
                       __hip_bfloat16* __restrict__ ygbf) {
    int d = blockIdx.x * blockDim.x + threadIdx.x;  // 0..DI-1
    int b = blockIdx.y;
    float A[NN];
#pragma unroll
    for (int n = 0; n < NN; ++n) A[n] = -__expf(Alog[(long)d * NN + n]);
    float Dv = Dvec[d];
    float h[NN];
#pragma unroll
    for (int n = 0; n < NN; ++n) h[n] = 0.0f;

    __shared__ float sBC[2 * NN];
    for (int t = 0; t < LL; ++t) {
        long bl = (long)b * LL + t;
        if (threadIdx.x < 2 * NN)
            sBC[threadIdx.x] = proj[bl * PROJW + RR + threadIdx.x];
        __syncthreads();
        float dtv = dt[bl * DI + d];
        float u   = xc[bl * DI + d];
        float zz  = xz[bl * XZW + DI + d];
        float y = 0.0f;
#pragma unroll
        for (int n = 0; n < NN; ++n) {
            h[n] = __expf(dtv * A[n]) * h[n] + dtv * sBC[n] * u;
            y = fmaf(h[n], sBC[NN + n], y);
        }
        float yg = (y + Dv * u) * silu_f(zz);
        ygbf[bl * DI + d] = __float2bfloat16(yg);
        __syncthreads();
    }
}

// lb_loss, load, entropy into d_out tail.
__global__ void k_finalize(const float* __restrict__ load_acc,
                           const float* __restrict__ ent_acc,
                           float* __restrict__ out_tail) {
    if (threadIdx.x != 0 || blockIdx.x != 0) return;
    const float target = (float)(ML * TOPK) / EE;
    float ln[EE], mean = 0.0f;
#pragma unroll
    for (int e = 0; e < EE; ++e) {
        ln[e] = load_acc[e] / (target + 1e-8f);
        mean += ln[e] * (1.0f / EE);
    }
    float var = 0.0f;
#pragma unroll
    for (int e = 0; e < EE; ++e) var += (ln[e] - mean) * (ln[e] - mean);
    var /= (float)(EE - 1);
    float cv = sqrtf(var) / fmaxf(mean, 1e-8f);
    out_tail[0] = cv * 0.01f * EE;
#pragma unroll
    for (int e = 0; e < EE; ++e) out_tail[1 + e] = load_acc[e];
    out_tail[1 + EE] = ent_acc[0] / (float)ML;
}

// ---------------- host ----------------
static inline size_t align256(size_t x) { return (x + 255) & ~(size_t)255; }

extern "C" void kernel_launch(void* const* d_in, const int* in_sizes, int n_in,
                              void* d_out, int out_size, void* d_ws, size_t ws_size,
                              hipStream_t stream) {
    const float* d_x     = (const float*)d_in[0];
    const float* d_gatew = (const float*)d_in[1];
    const float* d_Win   = (const float*)d_in[2];
    const float* d_convw = (const float*)d_in[3];
    const float* d_convb = (const float*)d_in[4];
    const float* d_Wx    = (const float*)d_in[5];
    const float* d_Wdt   = (const float*)d_in[6];
    const float* d_bdt   = (const float*)d_in[7];
    const float* d_Alog  = (const float*)d_in[8];
    const float* d_Dvec  = (const float*)d_in[9];
    const float* d_Wout  = (const float*)d_in[10];
    float* out = (float*)d_out;

    // workspace layout
    char* base = (char*)d_ws;
    size_t off = 0;
    auto take = [&](size_t bytes) { char* p = base + off; off += align256(bytes); return p; };
    __hip_bfloat16* xbf    = (__hip_bfloat16*)take((size_t)ML * DD * 2);
    __hip_bfloat16* w1bf   = (__hip_bfloat16*)take((size_t)XZW * DD * 2);
    __hip_bfloat16* wxbf   = (__hip_bfloat16*)take((size_t)PROJW * DI * 2);
    __hip_bfloat16* wdtbf  = (__hip_bfloat16*)take((size_t)DI * RR * 2);
    __hip_bfloat16* woutbf = (__hip_bfloat16*)take((size_t)DD * DI * 2);
    float* xz    = (float*)take((size_t)ML * XZW * 4);
    float* xc    = (float*)take((size_t)ML * DI * 4);
    __hip_bfloat16* xcbf = (__hip_bfloat16*)take((size_t)ML * DI * 2);
    float* proj  = (float*)take((size_t)ML * PROJW * 4);
    __hip_bfloat16* dtin = (__hip_bfloat16*)take((size_t)ML * RR * 2);
    float* dt    = (float*)take((size_t)ML * DI * 4);
    __hip_bfloat16* ygbf = (__hip_bfloat16*)take((size_t)ML * DI * 2);
    float* weights = (float*)take((size_t)ML * EE * 4);
    float* stats   = (float*)take(16 * 4);   // [0..7] load, [8] entropy sum

    const long outN = (long)ML * DD;

    // zero output accumulator + stats
    k_zero<<<dim3((unsigned)((outN + 255) / 256)), 256, 0, stream>>>(out, outN);
    k_zero<<<1, 32, 0, stream>>>(stats, 9);

    // x -> bf16
    {
        long n = (long)ML * DD;
        k_f32_to_bf16<<<dim3((unsigned)((n + 255) / 256)), 256, 0, stream>>>(d_x, xbf, n);
    }

    // router
    k_router<<<dim3((ML + 127) / 128), 128, 0, stream>>>(d_x, d_gatew, weights,
                                                         stats, stats + EE);

    for (int e = 0; e < EE; ++e) {
        // weight conversions
        { long n = (long)XZW * DD;
          k_f32_to_bf16<<<dim3((unsigned)((n + 255) / 256)), 256, 0, stream>>>(
              d_Win + (size_t)e * n, w1bf, n); }
        { long n = (long)PROJW * DI;
          k_f32_to_bf16<<<dim3((unsigned)((n + 255) / 256)), 256, 0, stream>>>(
              d_Wx + (size_t)e * n, wxbf, n); }
        { long n = (long)DI * RR;
          k_f32_to_bf16<<<dim3((unsigned)((n + 255) / 256)), 256, 0, stream>>>(
              d_Wdt + (size_t)e * n, wdtbf, n); }
        { long n = (long)DD * DI;
          k_f32_to_bf16<<<dim3((unsigned)((n + 255) / 256)), 256, 0, stream>>>(
              d_Wout + (size_t)e * n, woutbf, n); }

        // GEMM1: xz = x @ Win^T   [2048 x 4096, K=1024]  (full tiles)
        k_wmma_gemm_full<<<dim3(ML / 256, XZW / 128), 256, 0, stream>>>(
            xbf, w1bf, xz, ML, XZW, DD, 0, nullptr, nullptr, 0);

        // conv + silu
        { long n = (long)ML * DI;
          k_conv_silu<<<dim3((unsigned)((n + 255) / 256)), 256, 0, stream>>>(
              xz, d_convw + (size_t)e * DI * KW, d_convb + (size_t)e * DI, xc, xcbf); }

        // GEMM2: proj = xc @ Wx^T   [2048 x 96, K=2048]  (guarded, N not /128)
        k_wmma_gemm_guard<<<dim3(ML / 128, (PROJW + 63) / 64), 128, 0, stream>>>(
            xcbf, wxbf, proj, ML, PROJW, DI);

        // dt_in slice -> bf16
        { long n = (long)ML * RR;
          k_slice_to_bf16<<<dim3((unsigned)((n + 255) / 256)), 256, 0, stream>>>(
              proj, dtin, ML, PROJW, RR); }

        // GEMM3: dt = softplus(dt_in @ Wdt^T + bdt)   [2048 x 2048, K=64]
        k_wmma_gemm_full<<<dim3(ML / 256, DI / 128), 256, 0, stream>>>(
            dtin, wdtbf, dt, ML, DI, RR, 1, d_bdt + (size_t)e * DI, nullptr, 0);

        // selective scan -> gated y (bf16)
        k_scan<<<dim3(DI / 256, BB), 256, 0, stream>>>(
            dt, proj, xc, xz, d_Alog + (size_t)e * DI * NN,
            d_Dvec + (size_t)e * DI, ygbf);

        // GEMM4: out += w_e * (yg @ Wout^T)   [2048 x 1024, K=2048]
        k_wmma_gemm_full<<<dim3(ML / 256, DD / 128), 256, 0, stream>>>(
            ygbf, woutbf, out, ML, DD, DI, 2, nullptr, weights + e, EE);
    }

    // lb_loss, load, entropy
    k_finalize<<<1, 32, 0, stream>>>(stats, stats + EE, out + outN);
    (void)in_sizes; (void)n_in; (void)out_size; (void)ws_size;
}
